// MoAKDALayer_27058293965528
// MI455X (gfx1250) — matrix operand order repaired and checked by reading, not compiled, not run on hardware
//
#include <hip/hip_runtime.h>
#include <hip/hip_bf16.h>
#include <hip/hip_fp16.h>

typedef _Float16 half_t;
typedef __attribute__((ext_vector_type(16))) _Float16 v16h;
typedef __attribute__((ext_vector_type(8)))  float    v8f;
typedef __attribute__((ext_vector_type(4)))  unsigned int u32x4;
typedef __attribute__((ext_vector_type(8)))  int      i32x8;
typedef __attribute__((ext_vector_type(4)))  int      i32x4;

#define B_      2
#define NM_     4
#define T_      2048
#define DIN_    256
#define ND_     1024          // NM*DIN
#define BT_     4096          // B*T
#define DK_     16
#define DV_     16
#define DK2_    32
#define EQ_     16
#define EK_     16
#define R_      4
#define DA_     25
#define DPG_    158
#define DPGP_   160           // padded to multiple of 32
#define WMHC_N  768           // 2 sets * 16 experts * (4+4+16)
#define RMS_EPS 1.1920929e-07f
#define TWO_PI  6.283185307179586f
#define LN10000_OVER_DK 0.5756462732485115f   // ln(10000)/16

enum {
  IN_STREAM=0, IN_WQ, IN_WK, IN_WV, IN_POPE_DELTA,
  IN_LORA_A_Q, IN_LORA_B_Q, IN_LORA_A_K, IN_LORA_B_K, IN_LORA_A_V, IN_LORA_B_V,
  IN_ALPHA_UP, IN_ALPHA_DOWN, IN_BETA_UP, IN_BETA_DOWN,
  IN_WPRE, IN_WO, IN_WPG1, IN_WPG2, IN_ROUTER_Q, IN_ROUTER_KV,
  IN_MQ_NORM, IN_MQ_PHI_PRE, IN_MQ_PHI_POST, IN_MQ_PHI_RES,
  IN_MQ_B_PRE, IN_MQ_B_POST, IN_MQ_B_RES, IN_MQ_A_PRE, IN_MQ_A_POST, IN_MQ_A_RES,
  IN_MK_NORM, IN_MK_PHI_PRE, IN_MK_PHI_POST, IN_MK_PHI_RES,
  IN_MK_B_PRE, IN_MK_B_POST, IN_MK_B_RES, IN_MK_A_PRE, IN_MK_A_POST, IN_MK_A_RES
};

__device__ inline float sigmoidf_(float x) { return 1.f / (1.f + __expf(-x)); }
__device__ inline float siluf_(float x)    { return x / (1.f + __expf(-x)); }
__device__ inline float softplusf_(float x){ return x > 20.f ? x : log1pf(__expf(x)); }
__device__ inline float pope_freq(int j)   { return __expf((float)j * LN10000_OVER_DK); }

// ---------------------------------------------------------------------------
// Tensor Data Mover: 2D tile load (32 rows x 32 f16) global -> LDS.
// D# built per CDNA5 ISA ch.8: group0 {count,lds_addr,global_addr,type=2},
// group1 {data_size=2B, tensor dims, tile 32x32, dim0 stride = K elements}.
// ---------------------------------------------------------------------------
__device__ inline void tdm_load_tile_a(const half_t* gsrc, void* lds_dst, int K, int M) {
  unsigned long long ga = (unsigned long long)(size_t)gsrc;
  unsigned lds = (unsigned)(size_t)lds_dst;       // flat LDS addr truncates to LDS offset
  u32x4 g0;
  g0.x = 1u;                                      // count=1, user descriptor
  g0.y = lds;                                     // lds_addr (bits 63:32)
  g0.z = (unsigned)ga;                            // global_addr[31:0]
  g0.w = (unsigned)((ga >> 32) & 0x1FFFFFFull) | 0x80000000u;  // addr[56:32] | type=2
  i32x8 g1;
  g1[0] = (int)(1u << 16);                        // data_size=1 -> 2 bytes/elem
  g1[1] = (int)((K & 0xFFFF) << 16);              // tensor_dim0[15:0]
  g1[2] = (int)(((K >> 16) & 0xFFFF) | ((M & 0xFFFF) << 16));   // dim0 hi | dim1 lo
  g1[3] = (int)(((M >> 16) & 0xFFFF) | (32u << 16));            // dim1 hi | tile_dim0=32
  g1[4] = 32;                                     // tile_dim1=32 rows (tile_dim2=0)
  g1[5] = K;                                      // tensor_dim0_stride[31:0]
  g1[6] = 0;
  g1[7] = 0;
  i32x4 z4 = {0, 0, 0, 0};
#if __clang_major__ >= 23
  i32x8 z8 = {0, 0, 0, 0, 0, 0, 0, 0};
  __builtin_amdgcn_tensor_load_to_lds(g0, g1, z4, z4, z8, 0);
#else
  __builtin_amdgcn_tensor_load_to_lds(g0, g1, z4, z4, 0);
#endif
}

// ---------------------------------------------------------------------------
// Wave32 WMMA GEMM: C(MxN,f32) = A(MxK,f16,row-major) * B (stored N x K, so a
// lane's 16 K-contiguous values are one 32B vector load).
// 128 threads = 4 waves. Block tile: 32(M) x 64(N); each wave: 2 M-tiles x
// 1 N-tile sharing one B fragment (2 v_wmma per K-step). A tiles are staged
// into double-buffered LDS by the Tensor Data Mover, pipelined one K-step
// ahead: TENSORcnt completes in-order per wave, so s_wait_tensorcnt(1) after
// issuing tile i+1 guarantees tile i has landed. M,K mult of 32, N mult of 16.
// ---------------------------------------------------------------------------
__global__ __launch_bounds__(128)
void gemm_f16_wmma_kernel(const half_t* __restrict__ A, const half_t* __restrict__ Bt,
                          float* __restrict__ C, int M, int N, int K,
                          long sA, long sB, long sC) {
  const half_t* Ab = A  + (long)blockIdx.z * sA;
  const half_t* Bb = Bt + (long)blockIdx.z * sB;
  float*        Cb = C  + (long)blockIdx.z * sC;

  const int wave = threadIdx.x >> 5;
  const int lane = threadIdx.x & 31;
  const int m0 = blockIdx.y * 32;
  const int n0 = (blockIdx.x * 4 + wave) * 16;
  const bool nvalid = (n0 < N);

  __shared__ half_t As[2][32 * 32];

  v8f acc0 = {}, acc1 = {};
  const int mA = lane & 15;   // A fragment row within 16-row tile
  const int kh = lane >> 4;   // A K-half select
  const int nB = lane & 15;   // B fragment col
  const int ks = lane >> 4;   // B K-section select

  const int kiters = K >> 5;
  if (wave == 0)              // prologue: fill buffer 0 (TDM ignores EXEC)
    tdm_load_tile_a(Ab + (long)m0 * K, (void*)As[0], K, M);

  for (int i = 0; i < kiters; ++i) {
    const int k0 = i << 5;
    if (wave == 0) {
      if (i + 1 < kiters) {   // issue next tile into the other buffer, then
        tdm_load_tile_a(Ab + (long)m0 * K + k0 + 32, (void*)As[(i + 1) & 1], K, M);
        __builtin_amdgcn_s_wait_tensorcnt(1);   // wait for tile i only
      } else {
        __builtin_amdgcn_s_wait_tensorcnt(0);
      }
    }
    __syncthreads();
    if (nvalid) {
      const half_t* Al = As[i & 1];
      v16h bf = *(const v16h*)(Bb + (long)(n0 + nB) * K + k0 + ks * 16);
      v16h a0, a1;
#pragma unroll
      for (int j = 0; j < 8; ++j) {
        a0[j]     = Al[mA * 32 + kh * 16 + j];
        a0[8 + j] = Al[mA * 32 + kh * 16 + 8 + j];
        a1[j]     = Al[(16 + mA) * 32 + kh * 16 + j];
        a1[8 + j] = Al[(16 + mA) * 32 + kh * 16 + 8 + j];
      }
      acc0 = __builtin_amdgcn_wmma_f32_16x16x32_f16(false, a0, false, bf,
                                                    (short)0, acc0, false, false);
      acc1 = __builtin_amdgcn_wmma_f32_16x16x32_f16(false, a1, false, bf,
                                                    (short)0, acc1, false, false);
      if (k0 + 32 < K)
        __builtin_prefetch(Bb + (long)(n0 + nB) * K + k0 + 32 + ks * 16, 0, 2);
    }
    __syncthreads();
  }

  if (nvalid) {
    const int row = m0 + (lane >> 4) * 8;
    const int col = n0 + (lane & 15);
#pragma unroll
    for (int r = 0; r < 8; ++r) {
      Cb[(long)(row + r) * N + col]      = acc0[r];
      Cb[(long)(row + 16 + r) * N + col] = acc1[r];
    }
  }
}

// ---------------------------------------------------------------------------
// Weight preparation kernels (all B matrices stored transposed: N x K)
// ---------------------------------------------------------------------------
__global__ void build_wmhc_kernel(const float* nq, const float* ppq, const float* poq,
                                  const float* prq, const float* nk, const float* ppk,
                                  const float* pok, const float* prk, half_t* W) {
  int idx = blockIdx.x * blockDim.x + threadIdx.x;   // c * ND_ + row
  if (idx >= ND_ * WMHC_N) return;
  int c = idx / ND_, row = idx % ND_;
  int set = c / 384, ce = c % 384;
  int e = ce / 24, j = ce % 24;
  float v;
  if (j < 4)      v = (set ? ppk : ppq)[((long)e * ND_ + row) * 4 + j];
  else if (j < 8) v = (set ? pok : poq)[((long)e * ND_ + row) * 4 + (j - 4)];
  else            v = (set ? prk : prq)[((long)e * ND_ + row) * 16 + (j - 8)];
  v *= (set ? nk : nq)[(long)e * ND_ + row];
  W[idx] = (half_t)v;
}

__global__ void build_wcat_q_kernel(const float* Wq, const float* loraAq, half_t* out) {
  int idx = blockIdx.x * blockDim.x + threadIdx.x;   // e*32*DIN_ + j*DIN_ + row
  if (idx >= EQ_ * 32 * DIN_) return;
  int e = idx / (32 * DIN_), rem = idx % (32 * DIN_);
  int j = rem / DIN_, row = rem % DIN_;
  float v = 0.f;
  if (j < 16)      v = Wq[row * 16 + j];
  else if (j < 20) v = loraAq[((long)e * DIN_ + row) * R_ + (j - 16)];
  out[idx] = (half_t)v;
}

__global__ void build_wcat_kv_kernel(const float* Wk, const float* Wv,
                                     const float* loraAk, const float* loraAv,
                                     const float* aUp, const float* bUp, half_t* out) {
  int idx = blockIdx.x * blockDim.x + threadIdx.x;   // e*96*DIN_ + j*DIN_ + row
  if (idx >= EK_ * 96 * DIN_) return;
  int e = idx / (96 * DIN_), rem = idx % (96 * DIN_);
  int j = rem / DIN_, row = rem % DIN_;
  float v = 0.f;
  if (j < 16)      v = Wk[row * 16 + j];
  else if (j < 32) v = Wv[row * 16 + (j - 16)];
  else if (j < 36) v = loraAk[((long)e * DIN_ + row) * R_ + (j - 32)];
  else if (j < 40) v = loraAv[((long)e * DIN_ + row) * R_ + (j - 36)];
  else if (j < 65) v = aUp[((long)e * DIN_ + row) * DA_ + (j - 40)];
  else if (j < 90) v = bUp[((long)e * DIN_ + row) * DA_ + (j - 65)];
  out[idx] = (half_t)v;
}

// dst (N x K) = transpose of src (K x N), cast to f16
__global__ void transpose_cast_kernel(const float* src, half_t* dst, int Kdim, int Ndim) {
  int idx = blockIdx.x * blockDim.x + threadIdx.x;
  if (idx >= Kdim * Ndim) return;
  int n = idx / Kdim, k = idx % Kdim;
  dst[idx] = (half_t)src[(long)k * Ndim + n];
}

__global__ void build_wpg1_kernel(const float* W, half_t* out) {   // (DPGP_ x DIN_)
  int idx = blockIdx.x * blockDim.x + threadIdx.x;
  if (idx >= DPGP_ * DIN_) return;
  int n = idx / DIN_, k = idx % DIN_;
  out[idx] = (half_t)(n < DPG_ ? W[(long)k * DPG_ + n] : 0.f);
}
__global__ void build_wpg2_kernel(const float* W, half_t* out) {   // (DIN_ x DPGP_)
  int idx = blockIdx.x * blockDim.x + threadIdx.x;
  if (idx >= DIN_ * DPGP_) return;
  int n = idx / DPGP_, k = idx % DPGP_;
  out[idx] = (half_t)(k < DPG_ ? W[(long)k * DIN_ + n] : 0.f);
}

// ---------------------------------------------------------------------------
// RMS-norm of the concatenated stream token vector (1024) -> f16 (norm_w folded
// into the mHC weight matrix, so not applied here).
// ---------------------------------------------------------------------------
__global__ __launch_bounds__(256)
void rmsnorm_kernel(const float* stream_in, half_t* xnorm) {
  int bt = blockIdx.x, d = threadIdx.x;
  int b = bt / T_, t = bt % T_;
  float x[NM_], ss = 0.f;
#pragma unroll
  for (int n = 0; n < NM_; ++n) {
    x[n] = stream_in[(((long)(b * NM_ + n)) * T_ + t) * DIN_ + d];
    ss += x[n] * x[n];
  }
  __shared__ float red[256];
  red[d] = ss;
  __syncthreads();
  for (int s = 128; s > 0; s >>= 1) {
    if (d < s) red[d] += red[d + s];
    __syncthreads();
  }
  float scale = rsqrtf(red[0] / (float)ND_ + RMS_EPS);
#pragma unroll
  for (int n = 0; n < NM_; ++n)
    xnorm[(long)bt * ND_ + n * DIN_ + d] = (half_t)(x[n] * scale);
}

// ---------------------------------------------------------------------------
// Routing: mean-over-streams, two 256x16 matvecs, softmax + top-prob-max-k.
// One wave32 per token; lanes 0..15 -> q columns, 16..31 -> kv columns.
// ---------------------------------------------------------------------------
__global__ __launch_bounds__(32)
void routing_kernel(const float* stream_in, const float* router_q,
                    const float* router_kv, float* gate_q, float* gate_kv) {
  int bt = blockIdx.x, lane = threadIdx.x;
  int b = bt / T_, t = bt % T_;
  __shared__ float lg[32];
  int col = lane & 15;
  const float* Rw = (lane < 16) ? router_q : router_kv;
  float acc = 0.f;
  for (int d = 0; d < DIN_; ++d) {
    float ri = 0.f;
#pragma unroll
    for (int n = 0; n < NM_; ++n)
      ri += stream_in[(((long)(b * NM_ + n)) * T_ + t) * DIN_ + d];
    acc += 0.25f * ri * Rw[d * 16 + col];
  }
  lg[lane] = acc;
  __syncthreads();
  if (lane == 0 || lane == 16) {
    const float* L = &lg[lane];
    float p[16];
    float mx = L[0];
    for (int i = 1; i < 16; ++i) mx = fmaxf(mx, L[i]);
    float s = 0.f;
    for (int i = 0; i < 16; ++i) { p[i] = __expf(L[i] - mx); s += p[i]; }
    float invs = 1.f / s;
    for (int i = 0; i < 16; ++i) p[i] *= invs;
    float sel[16]; bool used[16];
    for (int i = 0; i < 16; ++i) { sel[i] = 0.f; used[i] = false; }
    float cum = 0.f;
    for (int rank = 0; rank < 16; ++rank) {
      int best = 0; float bv = -1.f;
      for (int i = 0; i < 16; ++i)
        if (!used[i] && p[i] > bv) { bv = p[i]; best = i; }
      used[best] = true;
      if (rank == 0 || (cum < 0.8f && rank < 4)) sel[best] = bv;
      cum += bv;
    }
    float* g = (lane == 0) ? (gate_q + (long)bt * 16) : (gate_kv + (long)bt * 16);
    for (int i = 0; i < 16; ++i) g[i] = sel[i];
  }
}

// ---------------------------------------------------------------------------
// mHC activations: sigmoid gates + 6-iteration sinkhorn on the 4x4 block.
// raw points at this set's 384-column slice (row stride WMHC_N).
// ---------------------------------------------------------------------------
__global__ void mhc_act_kernel(const float* raw, const float* b_pre, const float* b_post,
                               const float* b_res, const float* a_pre, const float* a_post,
                               const float* a_res, float* Hpre, float* Hpost, float* Hres) {
  int idx = blockIdx.x * blockDim.x + threadIdx.x;
  if (idx >= EQ_ * BT_) return;
  int e = idx / BT_, bt = idx % BT_;
  const float* r = raw + (long)bt * WMHC_N + e * 24;
  float ap = a_pre[e], apo = a_post[e], ar = a_res[e];
  float* hp = Hpre + ((long)e * BT_ + bt) * 4;
  float* ho = Hpost + ((long)e * BT_ + bt) * 4;
  for (int n = 0; n < 4; ++n) {
    hp[n] = sigmoidf_(ap * r[n] + b_pre[e * 4 + n]);
    ho[n] = 2.f * sigmoidf_(apo * r[4 + n] + b_post[e * 4 + n]);
  }
  float Mx[16];
  for (int i = 0; i < 16; ++i) Mx[i] = __expf(ar * r[8 + i] + b_res[e * 16 + i]);
  for (int it = 0; it < 6; ++it) {
    for (int i = 0; i < 4; ++i) {
      float s = Mx[i*4] + Mx[i*4+1] + Mx[i*4+2] + Mx[i*4+3];
      float inv = 1.f / s;
      for (int j = 0; j < 4; ++j) Mx[i*4+j] *= inv;
    }
    for (int j = 0; j < 4; ++j) {
      float s = Mx[j] + Mx[4+j] + Mx[8+j] + Mx[12+j];
      float inv = 1.f / s;
      for (int i = 0; i < 4; ++i) Mx[i*4+j] *= inv;
    }
  }
  float* hr = Hres + ((long)e * BT_ + bt) * 16;
  for (int i = 0; i < 16; ++i) hr[i] = Mx[i];
}

// h_e[set][e][bt][d] = sum_n Hpre[e][bt][n] * stream[b][n][t][d]  (f16 out)
__global__ __launch_bounds__(256)
void compute_h_kernel(const float* stream_in, const float* Hpre_q, const float* Hpre_k,
                      half_t* h_q, half_t* h_kv) {
  int bt = blockIdx.x, e = blockIdx.y, set = blockIdx.z, d = threadIdx.x;
  int b = bt / T_, t = bt % T_;
  const float* Hp = (set == 0 ? Hpre_q : Hpre_k) + ((long)e * BT_ + bt) * 4;
  float acc = 0.f;
#pragma unroll
  for (int n = 0; n < NM_; ++n)
    acc += Hp[n] * stream_in[(((long)(b * NM_ + n)) * T_ + t) * DIN_ + d];
  half_t* out = (set == 0 ? h_q : h_kv);
  out[((long)e * BT_ + bt) * DIN_ + d] = (half_t)acc;
}

// ---------------------------------------------------------------------------
// Per-expert epilogues (LoRA-B, l2norm, pope, silu/sigmoid heads)
// ---------------------------------------------------------------------------
__global__ void epilogue_q_kernel(const float* pq_raw, const float* loraBq, float* q_e) {
  int idx = blockIdx.x * blockDim.x + threadIdx.x;
  if (idx >= EQ_ * BT_) return;
  int e = idx / BT_, bt = idx % BT_;
  int t = bt % T_;
  const float* r = pq_raw + ((long)e * BT_ + bt) * 32;
  float lr[4];
  for (int i = 0; i < 4; ++i) lr[i] = siluf_(r[16 + i]);
  float qv[16], ss = 0.f;
  for (int j = 0; j < 16; ++j) {
    float dq = 0.f;
    for (int i = 0; i < 4; ++i) dq += lr[i] * loraBq[e * 64 + i * 16 + j];
    float x = r[j] + dq;
    qv[j] = x; ss += x * x;
  }
  float inv = 1.f / fmaxf(sqrtf(ss), 1e-12f);
  float* o = q_e + ((long)e * BT_ + bt) * 32;
  for (int j = 0; j < 16; ++j) {
    float mu = softplusf_(qv[j] * inv);
    float phi = (float)t * pope_freq(j);
    float sn, cs; __sincosf(phi, &sn, &cs);
    o[j] = mu * cs; o[16 + j] = mu * sn;
  }
}

__global__ void epilogue_kv_kernel(const float* pkv_raw, const float* pope_delta,
                                   const float* loraBk, const float* loraBv,
                                   const float* alphaDown, const float* betaDown,
                                   float* k_e, float* v_e, float* alpha_e, float* beta_e) {
  int idx = blockIdx.x * blockDim.x + threadIdx.x;
  if (idx >= EK_ * BT_) return;
  int e = idx / BT_, bt = idx % BT_;
  int t = bt % T_;
  const float* r = pkv_raw + ((long)e * BT_ + bt) * 96;
  float lk[4], lv[4];
  for (int i = 0; i < 4; ++i) { lk[i] = siluf_(r[32 + i]); lv[i] = siluf_(r[36 + i]); }
  float kv_[16], ss = 0.f;
  for (int j = 0; j < 16; ++j) {
    float dk = 0.f, dv = 0.f;
    for (int i = 0; i < 4; ++i) {
      dk += lk[i] * loraBk[e * 64 + i * 16 + j];
      dv += lv[i] * loraBv[e * 64 + i * 16 + j];
    }
    float kx = r[j] + dk;
    kv_[j] = kx; ss += kx * kx;
    v_e[((long)e * BT_ + bt) * 16 + j] = siluf_(r[16 + j] + dv);
  }
  float inv = 1.f / fmaxf(sqrtf(ss), 1e-12f);
  float* ko = k_e + ((long)e * BT_ + bt) * 32;
  for (int j = 0; j < 16; ++j) {
    float mu = softplusf_(kv_[j] * inv);
    float phi = (float)t * pope_freq(j) - TWO_PI * sigmoidf_(pope_delta[j]);
    float sn, cs; __sincosf(phi, &sn, &cs);
    ko[j] = mu * cs; ko[16 + j] = mu * sn;
  }
  float ar[DA_];
  for (int i = 0; i < DA_; ++i) ar[i] = siluf_(r[40 + i]);
  float* ao = alpha_e + ((long)e * BT_ + bt) * 32;
  for (int j = 0; j < 32; ++j) {
    float s2 = 0.f;
    for (int i = 0; i < DA_; ++i) s2 += ar[i] * alphaDown[e * DA_ * 32 + i * 32 + j];
    ao[j] = sigmoidf_(s2);
  }
  float bsum = 0.f;
  for (int i = 0; i < DA_; ++i) bsum += siluf_(r[65 + i]) * betaDown[e * DA_ + i];
  beta_e[(long)e * BT_ + bt] = sigmoidf_(bsum);
}

// ---------------------------------------------------------------------------
// Gate-weighted sums over experts
// ---------------------------------------------------------------------------
__global__ __launch_bounds__(256)
void route_combine_kernel(const float* gate_q, const float* gate_kv,
                          const half_t* h_q, const half_t* h_kv,
                          const float* q_e, const float* k_e, const float* v_e,
                          const float* alpha_e, const float* beta_e,
                          half_t* hq_rout, half_t* hkv_rout,
                          float* q_total, float* k_total, float* v_total,
                          float* alpha_total, float* beta_total) {
  int bt = blockIdx.x, d = threadIdx.x;
  __shared__ float gq[16], gk[16];
  if (d < 16) gq[d] = gate_q[(long)bt * 16 + d];
  else if (d < 32) gk[d - 16] = gate_kv[(long)bt * 16 + (d - 16)];
  __syncthreads();
  float aq = 0.f, ak = 0.f;
  for (int e = 0; e < 16; ++e) {
    aq += gq[e] * (float)h_q[((long)e * BT_ + bt) * DIN_ + d];
    ak += gk[e] * (float)h_kv[((long)e * BT_ + bt) * DIN_ + d];
  }
  hq_rout[(long)bt * DIN_ + d]  = (half_t)aq;
  hkv_rout[(long)bt * DIN_ + d] = (half_t)ak;
  if (d < 32) {
    float qt = 0.f, kt = 0.f, at = 0.f;
    for (int e = 0; e < 16; ++e) {
      qt += gq[e] * q_e[((long)e * BT_ + bt) * 32 + d];
      kt += gk[e] * k_e[((long)e * BT_ + bt) * 32 + d];
      at += gk[e] * alpha_e[((long)e * BT_ + bt) * 32 + d];
    }
    q_total[(long)bt * 32 + d] = qt;
    k_total[(long)bt * 32 + d] = kt;
    alpha_total[(long)bt * 32 + d] = at;
  }
  if (d < 16) {
    float vt = 0.f;
    for (int e = 0; e < 16; ++e) vt += gk[e] * v_e[((long)e * BT_ + bt) * 16 + d];
    v_total[(long)bt * 16 + d] = vt;
  }
  if (d == 0) {
    float btot = 0.f;
    for (int e = 0; e < 16; ++e) btot += gk[e] * beta_e[(long)e * BT_ + bt];
    beta_total[bt] = btot;
  }
}

// ---------------------------------------------------------------------------
// Serial KDA scan: 1 block per batch, 16 wave32s (one per DV column e),
// lane = DK2 index d. State S[d][e] lives in a register; wave shuffles reduce.
// ---------------------------------------------------------------------------
__global__ __launch_bounds__(512)
void kda_scan_kernel(const float* q_total, const float* k_total, const float* v_total,
                     const float* alpha_total, const float* beta_total,
                     float* scan_out, float* S_out) {
  int b = blockIdx.x;
  int e = threadIdx.x >> 5;
  int d = threadIdx.x & 31;
  float S = 0.f;
  for (int t = 0; t < T_; ++t) {
    long bt = (long)b * T_ + t;
    float kd = k_total[bt * 32 + d];
    float ad = alpha_total[bt * 32 + d];
    float qd = q_total[bt * 32 + d];
    float ve = v_total[bt * 16 + e];
    float be = beta_total[bt];
    float aS = ad * S;
    float x = kd * aS;
#pragma unroll
    for (int off = 16; off > 0; off >>= 1) x += __shfl_xor(x, off, 32);
    S = aS + be * kd * (ve - x);
    float o = qd * S;
#pragma unroll
    for (int off = 16; off > 0; off >>= 1) o += __shfl_xor(o, off, 32);
    if (d == 0) scan_out[bt * 16 + e] = o;
  }
  S_out[(long)b * (DK2_ * DV_) + d * 16 + e] = S;   // S_new (B, 32, 16)
}

// ---------------------------------------------------------------------------
// Output path fusions
// ---------------------------------------------------------------------------
__global__ __launch_bounds__(256)
void make_hp_kernel(const float* pre_raw, const float* scan_out, const float* gate_kv,
                    half_t* hp) {
  int bt = blockIdx.x, d = threadIdx.x;
  float pg = siluf_(pre_raw[(long)bt * DIN_ + d]);
  float head = scan_out[(long)bt * 16 + (d & 15)] * gate_kv[(long)bt * 16 + (d >> 4)];
  hp[(long)bt * DIN_ + d] = (half_t)(pg * head);
}

__global__ void silu_cast_kernel(const float* src, half_t* dst, int n) {
  int i = blockIdx.x * blockDim.x + threadIdx.x;
  if (i < n) dst[i] = (half_t)siluf_(src[i]);
}

__global__ void result_kernel(const float* proj, const float* pg2, float* result, int n) {
  int i = blockIdx.x * blockDim.x + threadIdx.x;
  if (i < n) result[i] = proj[i] * sigmoidf_(pg2[i]);
}

__global__ void combine_H_kernel(const float* gate_q, const float* gate_kv,
                                 const float* Hres_q, const float* Hres_k,
                                 const float* Hpost_q, const float* Hpost_k,
                                 float* Hres_comb, float* Hpost_comb) {
  int bt = blockIdx.x * blockDim.x + threadIdx.x;
  if (bt >= BT_) return;
  float hc[16], pc[4];
  for (int i = 0; i < 16; ++i) hc[i] = 0.f;
  for (int i = 0; i < 4; ++i)  pc[i] = 0.f;
  for (int e = 0; e < 16; ++e) {
    float gq = gate_q[(long)bt * 16 + e];
    float gk = gate_kv[(long)bt * 16 + e];
    const float* rq = Hres_q + ((long)e * BT_ + bt) * 16;
    const float* rk = Hres_k + ((long)e * BT_ + bt) * 16;
    for (int i = 0; i < 16; ++i) hc[i] += gq * rq[i] + gk * rk[i];
    const float* pq = Hpost_q + ((long)e * BT_ + bt) * 4;
    const float* pk = Hpost_k + ((long)e * BT_ + bt) * 4;
    for (int i = 0; i < 4; ++i) pc[i] += gq * pq[i] + gk * pk[i];
  }
  for (int i = 0; i < 16; ++i) Hres_comb[(long)bt * 16 + i] = hc[i];
  for (int i = 0; i < 4; ++i)  Hpost_comb[(long)bt * 4 + i] = pc[i];
}

__global__ __launch_bounds__(256)
void final_update_kernel(const float* stream_in, const float* Hres_comb,
                         const float* Hpost_comb, const float* result, float* out) {
  int bt = blockIdx.x, i = blockIdx.y, d = threadIdx.x;
  int b = bt / T_, t = bt % T_;
  float acc = Hpost_comb[(long)bt * 4 + i] * result[(long)bt * DIN_ + d];
#pragma unroll
  for (int j = 0; j < NM_; ++j)
    acc += Hres_comb[(long)bt * 16 + i * 4 + j] *
           stream_in[(((long)(b * NM_ + j)) * T_ + t) * DIN_ + d];
  out[(((long)(b * NM_ + i)) * T_ + t) * DIN_ + d] = acc;
}

// ---------------------------------------------------------------------------
// Host: workspace carving + launch graph
// ---------------------------------------------------------------------------
extern "C" void kernel_launch(void* const* d_in, const int* in_sizes, int n_in,
                              void* d_out, int out_size, void* d_ws, size_t ws_size,
                              hipStream_t stream) {
  (void)in_sizes; (void)n_in; (void)out_size; (void)ws_size;
  const float* s_in   = (const float*)d_in[IN_STREAM];
  const float* Wq     = (const float*)d_in[IN_WQ];
  const float* Wk     = (const float*)d_in[IN_WK];
  const float* Wv     = (const float*)d_in[IN_WV];
  const float* pope_d = (const float*)d_in[IN_POPE_DELTA];
  const float* lAq    = (const float*)d_in[IN_LORA_A_Q];
  const float* lBq    = (const float*)d_in[IN_LORA_B_Q];
  const float* lAk    = (const float*)d_in[IN_LORA_A_K];
  const float* lBk    = (const float*)d_in[IN_LORA_B_K];
  const float* lAv    = (const float*)d_in[IN_LORA_A_V];
  const float* lBv    = (const float*)d_in[IN_LORA_B_V];
  const float* aUp    = (const float*)d_in[IN_ALPHA_UP];
  const float* aDown  = (const float*)d_in[IN_ALPHA_DOWN];
  const float* bUp    = (const float*)d_in[IN_BETA_UP];
  const float* bDown  = (const float*)d_in[IN_BETA_DOWN];
  const float* Wpre   = (const float*)d_in[IN_WPRE];
  const float* Wo     = (const float*)d_in[IN_WO];
  const float* Wpg1   = (const float*)d_in[IN_WPG1];
  const float* Wpg2   = (const float*)d_in[IN_WPG2];
  const float* rQ     = (const float*)d_in[IN_ROUTER_Q];
  const float* rKV    = (const float*)d_in[IN_ROUTER_KV];

  char* w = (char*)d_ws;
  auto alloc = [&](size_t bytes) -> void* {
    void* p = (void*)w;
    w += (bytes + 255) & ~(size_t)255;
    return p;
  };

  half_t* xnorm    = (half_t*)alloc((size_t)BT_ * ND_ * 2);
  half_t* Wmhc     = (half_t*)alloc((size_t)ND_ * WMHC_N * 2);
  float*  mhc_raw  = (float*) alloc((size_t)BT_ * WMHC_N * 4);
  float*  Hpre_q   = (float*) alloc((size_t)EQ_ * BT_ * 4 * 4);
  float*  Hpost_q  = (float*) alloc((size_t)EQ_ * BT_ * 4 * 4);
  float*  Hres_q   = (float*) alloc((size_t)EQ_ * BT_ * 16 * 4);
  float*  Hpre_k   = (float*) alloc((size_t)EK_ * BT_ * 4 * 4);
  float*  Hpost_k  = (float*) alloc((size_t)EK_ * BT_ * 4 * 4);
  float*  Hres_k   = (float*) alloc((size_t)EK_ * BT_ * 16 * 4);
  half_t* h_q      = (half_t*)alloc((size_t)EQ_ * BT_ * DIN_ * 2);
  half_t* h_kv     = (half_t*)alloc((size_t)EK_ * BT_ * DIN_ * 2);
  half_t* Wcat_q   = (half_t*)alloc((size_t)EQ_ * DIN_ * 32 * 2);
  half_t* Wcat_kv  = (half_t*)alloc((size_t)EK_ * DIN_ * 96 * 2);
  float*  pq_raw   = (float*) alloc((size_t)EQ_ * BT_ * 32 * 4);
  float*  pkv_raw  = (float*) alloc((size_t)EK_ * BT_ * 96 * 4);
  float*  q_e      = (float*) alloc((size_t)EQ_ * BT_ * 32 * 4);
  float*  k_e      = (float*) alloc((size_t)EK_ * BT_ * 32 * 4);
  float*  v_e      = (float*) alloc((size_t)EK_ * BT_ * 16 * 4);
  float*  alpha_e  = (float*) alloc((size_t)EK_ * BT_ * 32 * 4);
  float*  beta_e   = (float*) alloc((size_t)EK_ * BT_ * 4);
  float*  gate_q   = (float*) alloc((size_t)BT_ * 16 * 4);
  float*  gate_kv  = (float*) alloc((size_t)BT_ * 16 * 4);
  float*  q_total  = (float*) alloc((size_t)BT_ * 32 * 4);
  float*  k_total  = (float*) alloc((size_t)BT_ * 32 * 4);
  float*  v_total  = (float*) alloc((size_t)BT_ * 16 * 4);
  float*  a_total  = (float*) alloc((size_t)BT_ * 32 * 4);
  float*  b_total  = (float*) alloc((size_t)BT_ * 4);
  half_t* hq_rout  = (half_t*)alloc((size_t)BT_ * DIN_ * 2);
  half_t* hkv_rout = (half_t*)alloc((size_t)BT_ * DIN_ * 2);
  float*  scan_out = (float*) alloc((size_t)BT_ * 16 * 4);
  float*  pre_raw  = (float*) alloc((size_t)BT_ * DIN_ * 4);
  half_t* hp       = (half_t*)alloc((size_t)BT_ * DIN_ * 2);
  float*  proj     = (float*) alloc((size_t)BT_ * DIN_ * 4);
  float*  pg1_raw  = (float*) alloc((size_t)BT_ * DPGP_ * 4);
  half_t* pg1_h    = (half_t*)alloc((size_t)BT_ * DPGP_ * 2);
  float*  pg2_raw  = (float*) alloc((size_t)BT_ * DIN_ * 4);
  float*  result   = (float*) alloc((size_t)BT_ * DIN_ * 4);
  float*  HresC    = (float*) alloc((size_t)BT_ * 16 * 4);
  float*  HpostC   = (float*) alloc((size_t)BT_ * 4 * 4);
  half_t* Wpre_h   = (half_t*)alloc((size_t)DIN_ * DIN_ * 2);
  half_t* Wo_h     = (half_t*)alloc((size_t)DIN_ * DIN_ * 2);
  half_t* Wpg1_h   = (half_t*)alloc((size_t)DPGP_ * DIN_ * 2);
  half_t* Wpg2_h   = (half_t*)alloc((size_t)DIN_ * DPGP_ * 2);

  float* out_su = (float*)d_out;                         // (B,NM,T,DIN)
  float* out_S  = out_su + (size_t)B_ * NM_ * T_ * DIN_; // (B,32,16)

  // --- weight prep (all GEMM B-operands stored N x K) ---
  build_wmhc_kernel<<<(ND_ * WMHC_N + 255) / 256, 256, 0, stream>>>(
      (const float*)d_in[IN_MQ_NORM], (const float*)d_in[IN_MQ_PHI_PRE],
      (const float*)d_in[IN_MQ_PHI_POST], (const float*)d_in[IN_MQ_PHI_RES],
      (const float*)d_in[IN_MK_NORM], (const float*)d_in[IN_MK_PHI_PRE],
      (const float*)d_in[IN_MK_PHI_POST], (const float*)d_in[IN_MK_PHI_RES], Wmhc);
  build_wcat_q_kernel<<<(EQ_ * DIN_ * 32 + 255) / 256, 256, 0, stream>>>(Wq, lAq, Wcat_q);
  build_wcat_kv_kernel<<<(EK_ * DIN_ * 96 + 255) / 256, 256, 0, stream>>>(
      Wk, Wv, lAk, lAv, aUp, bUp, Wcat_kv);
  transpose_cast_kernel<<<(DIN_ * DIN_ + 255) / 256, 256, 0, stream>>>(
      Wpre, Wpre_h, DIN_, DIN_);
  transpose_cast_kernel<<<(DIN_ * DIN_ + 255) / 256, 256, 0, stream>>>(
      Wo, Wo_h, DIN_, DIN_);
  build_wpg1_kernel<<<(DPGP_ * DIN_ + 255) / 256, 256, 0, stream>>>(Wpg1, Wpg1_h);
  build_wpg2_kernel<<<(DIN_ * DPGP_ + 255) / 256, 256, 0, stream>>>(Wpg2, Wpg2_h);

  // --- activations prep ---
  rmsnorm_kernel<<<BT_, 256, 0, stream>>>(s_in, xnorm);
  routing_kernel<<<BT_, 32, 0, stream>>>(s_in, rQ, rKV, gate_q, gate_kv);

  // --- mHC projection GEMM: 4096 x 768 x 1024 ---
  gemm_f16_wmma_kernel<<<dim3(WMHC_N / 64, BT_ / 32, 1), 128, 0, stream>>>(
      xnorm, Wmhc, mhc_raw, BT_, WMHC_N, ND_, 0, 0, 0);

  mhc_act_kernel<<<(EQ_ * BT_ + 255) / 256, 256, 0, stream>>>(
      mhc_raw, (const float*)d_in[IN_MQ_B_PRE], (const float*)d_in[IN_MQ_B_POST],
      (const float*)d_in[IN_MQ_B_RES], (const float*)d_in[IN_MQ_A_PRE],
      (const float*)d_in[IN_MQ_A_POST], (const float*)d_in[IN_MQ_A_RES],
      Hpre_q, Hpost_q, Hres_q);
  mhc_act_kernel<<<(EK_ * BT_ + 255) / 256, 256, 0, stream>>>(
      mhc_raw + 384, (const float*)d_in[IN_MK_B_PRE], (const float*)d_in[IN_MK_B_POST],
      (const float*)d_in[IN_MK_B_RES], (const float*)d_in[IN_MK_A_PRE],
      (const float*)d_in[IN_MK_A_POST], (const float*)d_in[IN_MK_A_RES],
      Hpre_k, Hpost_k, Hres_k);

  compute_h_kernel<<<dim3(BT_, 16, 2), 256, 0, stream>>>(s_in, Hpre_q, Hpre_k, h_q, h_kv);

  // --- per-expert projections (batched over experts in blockIdx.z) ---
  gemm_f16_wmma_kernel<<<dim3(1, BT_ / 32, EQ_), 128, 0, stream>>>(
      h_q, Wcat_q, pq_raw, BT_, 32, DIN_,
      (long)BT_ * DIN_, (long)DIN_ * 32, (long)BT_ * 32);
  gemm_f16_wmma_kernel<<<dim3(2, BT_ / 32, EK_), 128, 0, stream>>>(
      h_kv, Wcat_kv, pkv_raw, BT_, 96, DIN_,
      (long)BT_ * DIN_, (long)DIN_ * 96, (long)BT_ * 96);

  epilogue_q_kernel<<<(EQ_ * BT_ + 255) / 256, 256, 0, stream>>>(pq_raw, lBq, q_e);
  epilogue_kv_kernel<<<(EK_ * BT_ + 255) / 256, 256, 0, stream>>>(
      pkv_raw, pope_d, lBk, lBv, aDown, bDown, k_e, v_e, alpha_e, beta_e);

  route_combine_kernel<<<BT_, 256, 0, stream>>>(
      gate_q, gate_kv, h_q, h_kv, q_e, k_e, v_e, alpha_e, beta_e,
      hq_rout, hkv_rout, q_total, k_total, v_total, a_total, b_total);

  // --- serial KDA scan (writes S_new tail of d_out) ---
  kda_scan_kernel<<<B_, 512, 0, stream>>>(
      q_total, k_total, v_total, a_total, b_total, scan_out, out_S);

  // --- output path ---
  gemm_f16_wmma_kernel<<<dim3(DIN_ / 64, BT_ / 32, 1), 128, 0, stream>>>(
      hkv_rout, Wpre_h, pre_raw, BT_, DIN_, DIN_, 0, 0, 0);
  make_hp_kernel<<<BT_, 256, 0, stream>>>(pre_raw, scan_out, gate_kv, hp);
  gemm_f16_wmma_kernel<<<dim3(DIN_ / 64, BT_ / 32, 1), 128, 0, stream>>>(
      hp, Wo_h, proj, BT_, DIN_, DIN_, 0, 0, 0);
  gemm_f16_wmma_kernel<<<dim3((DPGP_ / 16 + 3) / 4, BT_ / 32, 1), 128, 0, stream>>>(
      hq_rout, Wpg1_h, pg1_raw, BT_, DPGP_, DIN_, 0, 0, 0);
  silu_cast_kernel<<<(BT_ * DPGP_ + 255) / 256, 256, 0, stream>>>(
      pg1_raw, pg1_h, BT_ * DPGP_);
  gemm_f16_wmma_kernel<<<dim3(DIN_ / 64, BT_ / 32, 1), 128, 0, stream>>>(
      pg1_h, Wpg2_h, pg2_raw, BT_, DIN_, DPGP_, 0, 0, 0);
  result_kernel<<<(BT_ * DIN_ + 255) / 256, 256, 0, stream>>>(
      proj, pg2_raw, result, BT_ * DIN_);

  // --- final stream update ---
  combine_H_kernel<<<(BT_ + 127) / 128, 128, 0, stream>>>(
      gate_q, gate_kv, Hres_q, Hres_k, Hpost_q, Hpost_k, HresC, HpostC);
  final_update_kernel<<<dim3(BT_, NM_, 1), 256, 0, stream>>>(
      s_in, HresC, HpostC, result, out_su);
}
